// SlotAttention_10050223473012
// MI455X (gfx1250) — matrix operand order repaired and checked by reading, not compiled, not run on hardware
//
#include <hip/hip_runtime.h>
#include <stdint.h>

#define BATCH 32
#define NTOK  4096
#define DDIM  256
#define NSLOT 8
#define ITERS 3
#define HID   256
#define SCALEF 0.0625f   // 256^-0.5

typedef __attribute__((ext_vector_type(16))) __bf16 v16bf;
typedef __attribute__((ext_vector_type(8)))  float  v8f;

union FragU {
  v16bf v;
  uint4 q[2];
  unsigned short u[16];
};
union Pack8 {
  unsigned short u[8];
  uint4 q;
};

__device__ __forceinline__ unsigned short f2bf(float f) {
  unsigned u = __float_as_uint(f);
  u += 0x7fffu + ((u >> 16) & 1u);           // round-to-nearest-even
  return (unsigned short)(u >> 16);
}

// A-matrix 16x32 bf16 fragment from row-major storage: element (m,k) at base[m*ld+k].
// Lane m holds K={hi*8..hi*8+7, 16+hi*8..16+hi*8+7}  (hi = lane>>4), per ISA table.
__device__ __forceinline__ v16bf frag_a(const unsigned short* base, int ld, unsigned lane) {
  unsigned m = lane & 15u, hi = lane >> 4;
  const unsigned short* p = base + (size_t)m * ld + hi * 8;
  FragU f;
  f.q[0] = *(const uint4*)p;
  f.q[1] = *(const uint4*)(p + 16);
  return f.v;
}

// B-matrix 32x16 bf16 fragment from COLUMN-major storage: element (k,n) at base[n*ld+k].
// Lane = n; holds K={hi*8..hi*8+7, 16+hi*8..16+hi*8+7} (hi = lane>>4), matching frag_a.
// ("row striped across lanes within one VGPR" => lane indexes N).
__device__ __forceinline__ v16bf frag_b(const unsigned short* base, int ld, unsigned lane) {
  unsigned n = lane & 15u, hi = lane >> 4;
  const unsigned short* p = base + (size_t)n * ld + hi * 8;
  FragU f;
  f.q[0] = *(const uint4*)p;
  f.q[1] = *(const uint4*)(p + 16);
  return f.v;
}

__device__ __forceinline__ v8f wmma_bf16(v16bf a, v16bf b, v8f c) {
  return __builtin_amdgcn_wmma_f32_16x16x32_bf16(false, a, false, b, (short)0, c, false, false);
}

// ---------------------------------------------------------------- prep kernels
__global__ void prep_w_kernel(const float* __restrict__ Wk, const float* __restrict__ Wv,
                              unsigned short* __restrict__ Wkb, unsigned short* __restrict__ Wvb) {
  int idx = blockIdx.x * blockDim.x + threadIdx.x;     // 0..65535
  const float* src = blockIdx.y ? Wv : Wk;
  unsigned short* dst = blockIdx.y ? Wvb : Wkb;
  dst[idx] = f2bf(src[idx]);
}

__global__ void init_slots_kernel(const float* __restrict__ emb, float* __restrict__ slots) {
  int idx = blockIdx.x * blockDim.x + threadIdx.x;     // 0..65535
  slots[idx] = emb[idx & (NSLOT * DDIM - 1)];
}

// ------------------------------------------------- LN(input) + K/V projections
// One wave per 16-token tile.  K stored row-major bf16 [B,N,D] (A of dots GEMM),
// V stored transposed bf16 VT[B,D,N] (column-major B of updates GEMM).
__global__ void __launch_bounds__(128) ln_kv_kernel(
    const float* __restrict__ x, const float* __restrict__ lg, const float* __restrict__ lb,
    const unsigned short* __restrict__ Wkb, const unsigned short* __restrict__ Wvb,
    const float* __restrict__ bk, const float* __restrict__ bv,
    unsigned short* __restrict__ K, unsigned short* __restrict__ VT) {
  __shared__ __align__(16) unsigned short xln[4][16][DDIM];
  __shared__ __align__(16) unsigned short cstk[4][16][16];
  __shared__ __align__(16) unsigned short cstv[4][16][16];
  unsigned wave = threadIdx.x >> 5, lane = threadIdx.x & 31u;
  unsigned tile = blockIdx.x * 4u + wave;       // 0..8191
  unsigned bat = tile >> 8;                     // 256 tiles per batch
  unsigned j0  = (tile & 255u) * 16u;
  unsigned row = lane & 15u, half = lane >> 4;

  // layernorm over D=256: two lanes per row, 128 elems each
  const float* xr = x + ((size_t)tile * 16u + row) * DDIM + half * 128u;
  float s = 0.f, s2 = 0.f;
  for (int c = 0; c < 128; ++c) { float v = xr[c]; s += v; s2 += v * v; }
  s  += __shfl_xor(s, 16);
  s2 += __shfl_xor(s2, 16);
  float mu = s * (1.f / DDIM);
  float rs = rsqrtf(s2 * (1.f / DDIM) - mu * mu + 1e-5f);
  for (int c = 0; c < 128; ++c) {
    int d = half * 128 + c;
    xln[wave][row][d] = f2bf((xr[c] - mu) * rs * lg[d] + lb[d]);
  }
  __syncthreads();

  for (int nt = 0; nt < 16; ++nt) {             // 16 output-dim tiles
    v8f ck = {}; v8f cv = {};
    for (int kc = 0; kc < 8; ++kc) {            // K=256 in 32-chunks
      v16bf a   = frag_a(&xln[wave][0][0] + kc * 32, DDIM, lane);
      v16bf bkf = frag_b(Wkb + (size_t)nt * 16 * DDIM + kc * 32, DDIM, lane);
      v16bf bvf = frag_b(Wvb + (size_t)nt * 16 * DDIM + kc * 32, DDIM, lane);
      ck = wmma_bf16(a, bkf, ck);
      cv = wmma_bf16(a, bvf, cv);
    }
    float bks = bk[nt * 16 + (lane & 15u)];
    float bvs = bv[nt * 16 + (lane & 15u)];
#pragma unroll
    for (int r = 0; r < 8; ++r) {
      unsigned m = r + half * 8;                // C layout: VGPR r -> M=r (+8 for hi lanes)
      cstk[wave][m][lane & 15u] = f2bf(ck[r] + bks);
      cstv[wave][m][lane & 15u] = f2bf(cv[r] + bvs);
    }
    __syncthreads();
    if (lane < 16) {                            // K row-major, 32B per lane
      uint4* dst = (uint4*)(K + ((size_t)tile * 16u + lane) * DDIM + nt * 16);
      const uint4* src = (const uint4*)&cstk[wave][lane][0];
      dst[0] = src[0]; dst[1] = src[1];
    } else {                                    // V transposed: VT[b][d][j]
      unsigned n = lane - 16u;
      Pack8 p0, p1;
#pragma unroll
      for (int m = 0; m < 8; ++m) p0.u[m] = cstv[wave][m][n];
#pragma unroll
      for (int m = 0; m < 8; ++m) p1.u[m] = cstv[wave][m + 8][n];
      uint4* dst = (uint4*)(VT + ((size_t)bat * DDIM + nt * 16 + n) * NTOK + j0);
      dst[0] = p0.q; dst[1] = p1.q;
    }
    __syncthreads();
  }
}

// ---------------------------------------- per-iteration: LN(slots) + q, zero sums
__global__ void __launch_bounds__(256) slot_q_kernel(
    const float* __restrict__ slots, const float* __restrict__ lg, const float* __restrict__ lb,
    const float* __restrict__ Wq, const float* __restrict__ bq,
    unsigned short* __restrict__ qbf, float* __restrict__ sums) {
  __shared__ float sn[NSLOT][DDIM];
  int bat = blockIdx.x;
  unsigned wave = threadIdx.x >> 5, lane = threadIdx.x & 31u;
  const float* row = slots + ((size_t)bat * NSLOT + wave) * DDIM;   // wave = slot (8 waves)
  float s = 0.f, s2 = 0.f;
  for (int c = 0; c < 8; ++c) { float v = row[lane * 8 + c]; s += v; s2 += v * v; }
  for (int m = 16; m >= 1; m >>= 1) { s += __shfl_xor(s, m); s2 += __shfl_xor(s2, m); }
  float mu = s * (1.f / DDIM);
  float rs = rsqrtf(s2 * (1.f / DDIM) - mu * mu + 1e-5f);
  for (int c = 0; c < 8; ++c) {
    int d = lane * 8 + c;
    sn[wave][d] = (row[d] - mu) * rs * lg[d] + lb[d];
  }
  __syncthreads();
  int t = threadIdx.x;
  float acc[NSLOT];
#pragma unroll
  for (int i = 0; i < NSLOT; ++i) acc[i] = 0.f;
  for (int d = 0; d < DDIM; ++d) {
    float w = Wq[(size_t)t * DDIM + d];
#pragma unroll
    for (int i = 0; i < NSLOT; ++i) acc[i] += sn[i][d] * w;
  }
  float bqt = bq[t];
#pragma unroll
  for (int i = 0; i < NSLOT; ++i)
    qbf[((size_t)bat * 16u + i) * DDIM + t] = f2bf(acc[i] + bqt);
#pragma unroll
  for (int i = NSLOT; i < 16; ++i)            // zero padding slot rows
    qbf[((size_t)bat * 16u + i) * DDIM + t] = 0;
  if (t < 16) sums[bat * 16 + t] = 0.f;
}

// ------------------- dots^T = K @ q^T, softmax over slots, partial j-sums, attn
__global__ void __launch_bounds__(128) dots_attn_kernel(
    const unsigned short* __restrict__ K, const unsigned short* __restrict__ qbf,
    unsigned short* __restrict__ attn, float* __restrict__ sums) {
  unsigned wave = threadIdx.x >> 5, lane = threadIdx.x & 31u;
  unsigned gid = blockIdx.x * 4u + wave;          // 0..8191
  unsigned bat = gid >> 8, jt = gid & 255u;
  const unsigned short* Kb = K + ((size_t)bat * NTOK + jt * 16u) * DDIM;
  const unsigned short* qb = qbf + (size_t)bat * 16u * DDIM;
  v8f c = {};
  for (int kc = 0; kc < 8; ++kc) {
    v16bf a  = frag_a(Kb + kc * 32, DDIM, lane);  // A: 16 tokens x 32 d
    v16bf bb = frag_b(qb + kc * 32, DDIM, lane);  // B: col-major = q row-major [i][d]
    c = wmma_bf16(a, bb, c);
  }
  // C[m=j][n=i]; each 16-lane group holds all 16 slot values of one j per register.
  unsigned i = lane & 15u;
  bool real = (i < NSLOT);
  float p[8]; float part = 0.f;
#pragma unroll
  for (int r = 0; r < 8; ++r) {
    float v = real ? c[r] * SCALEF : -__builtin_inff();
    float m = v;
    m = fmaxf(m, __shfl_xor(m, 8));
    m = fmaxf(m, __shfl_xor(m, 4));
    m = fmaxf(m, __shfl_xor(m, 2));
    m = fmaxf(m, __shfl_xor(m, 1));
    float e = real ? __expf(v - m) : 0.f;
    float se = e;
    se += __shfl_xor(se, 8);
    se += __shfl_xor(se, 4);
    se += __shfl_xor(se, 2);
    se += __shfl_xor(se, 1);
    float pv = real ? (e / se + 1e-8f) : 0.f;     // softmax over slots + EPS
    p[r] = pv; part += pv;
  }
  part += __shfl_xor(part, 16);                   // combine j=r and j=r+8 halves
  if (lane < 16u) atomicAdd(&sums[bat * 16u + i], part);
  Pack8 pk;
#pragma unroll
  for (int r = 0; r < 8; ++r) pk.u[r] = f2bf(p[r]);
  unsigned jo = jt * 16u + ((lane >= 16u) ? 8u : 0u);
  *(uint4*)(attn + ((size_t)bat * 16u + i) * NTOK + jo) = pk.q;   // rows 8..15 get zeros
}

// -------------------- updates = (attn/rowsum) @ V   via  attn[16xN] x VT tiles
__global__ void __launch_bounds__(128) updates_kernel(
    const unsigned short* __restrict__ attn, const unsigned short* __restrict__ VT,
    const float* __restrict__ sums, float* __restrict__ upd) {
  __shared__ __align__(16) float stage[4][16][16];
  unsigned wave = threadIdx.x >> 5, lane = threadIdx.x & 31u;
  unsigned gid = blockIdx.x * 4u + wave;          // 0..511
  unsigned bat = gid >> 4, nt = gid & 15u;
  const unsigned short* Ab = attn + (size_t)bat * 16u * NTOK;
  const unsigned short* Bb = VT + ((size_t)bat * DDIM + nt * 16u) * NTOK;
  v8f c = {};
  for (int kc = 0; kc < 128; ++kc) {              // K = N = 4096
    v16bf a  = frag_a(Ab + kc * 32, NTOK, lane);
    v16bf bb = frag_b(Bb + kc * 32, NTOK, lane);
    c = wmma_bf16(a, bb, c);
  }
  unsigned n = lane & 15u, half = lane >> 4;
#pragma unroll
  for (int r = 0; r < 8; ++r) {
    unsigned m = r + half * 8;
    float inv = (m < NSLOT) ? 1.f / sums[bat * 16u + m] : 0.f;  // fold normalizer
    stage[wave][m][n] = c[r] * inv;
  }
  __syncthreads();
  if (lane < NSLOT) {
    float* dst = upd + ((size_t)bat * NSLOT + lane) * DDIM + nt * 16u;
    for (int t = 0; t < 16; ++t) dst[t] = stage[wave][lane][t];
  }
  __syncthreads();
}

// ---------------------------------- GRU cell + LN + MLP + residual (tiny, fp32)
__global__ void __launch_bounds__(256) gru_mlp_kernel(
    const float* __restrict__ slots_in, const float* __restrict__ upd,
    const float* __restrict__ W_ih, const float* __restrict__ b_ih,
    const float* __restrict__ W_hh, const float* __restrict__ b_hh,
    const float* __restrict__ lg, const float* __restrict__ lb,
    const float* __restrict__ W1, const float* __restrict__ b1,
    const float* __restrict__ W2, const float* __restrict__ b2,
    float* __restrict__ slots_out, float* __restrict__ out2) {
  __shared__ float sprev[NSLOT][DDIM];
  __shared__ float updL[NSLOT][DDIM];
  __shared__ float smid[NSLOT][DDIM];
  __shared__ float hbuf[NSLOT][DDIM];
  __shared__ float actv[NSLOT][DDIM];
  __shared__ float mu8[NSLOT], rs8[NSLOT];
  int bat = blockIdx.x; int t = threadIdx.x;
#pragma unroll
  for (int s = 0; s < NSLOT; ++s) {
    sprev[s][t] = slots_in[((size_t)bat * NSLOT + s) * DDIM + t];
    updL[s][t]  = upd[((size_t)bat * NSLOT + s) * DDIM + t];
  }
  __syncthreads();
  float ar[8] = {}, az[8] = {}, an[8] = {}, gr[8] = {}, gz[8] = {}, gn[8] = {};
  for (int d = 0; d < DDIM; ++d) {
    float wr = W_ih[(size_t)t * DDIM + d];
    float wz = W_ih[((size_t)DDIM + t) * DDIM + d];
    float wn = W_ih[((size_t)2 * DDIM + t) * DDIM + d];
    float vr = W_hh[(size_t)t * DDIM + d];
    float vz = W_hh[((size_t)DDIM + t) * DDIM + d];
    float vn = W_hh[((size_t)2 * DDIM + t) * DDIM + d];
#pragma unroll
    for (int s = 0; s < NSLOT; ++s) {
      float u = updL[s][d], h = sprev[s][d];
      ar[s] += u * wr; az[s] += u * wz; an[s] += u * wn;
      gr[s] += h * vr; gz[s] += h * vz; gn[s] += h * vn;
    }
  }
  float bir = b_ih[t], biz = b_ih[DDIM + t], bin = b_ih[2 * DDIM + t];
  float bhr = b_hh[t], bhz = b_hh[DDIM + t], bhn = b_hh[2 * DDIM + t];
#pragma unroll
  for (int s = 0; s < NSLOT; ++s) {
    float r = 1.f / (1.f + __expf(-(ar[s] + bir + gr[s] + bhr)));
    float z = 1.f / (1.f + __expf(-(az[s] + biz + gz[s] + bhz)));
    float nn = tanhf(an[s] + bin + r * (gn[s] + bhn));
    smid[s][t] = (1.f - z) * nn + z * sprev[s][t];
  }
  __syncthreads();
  unsigned wave = t >> 5, lane = t & 31u;         // 8 waves = 8 slot rows
  {
    float s1 = 0.f, s2 = 0.f;
    for (int c = 0; c < 8; ++c) { float v = smid[wave][lane * 8 + c]; s1 += v; s2 += v * v; }
    for (int m = 16; m >= 1; m >>= 1) { s1 += __shfl_xor(s1, m); s2 += __shfl_xor(s2, m); }
    if (lane == 0) {
      float mu = s1 * (1.f / DDIM);
      mu8[wave] = mu;
      rs8[wave] = rsqrtf(s2 * (1.f / DDIM) - mu * mu + 1e-5f);
    }
  }
  __syncthreads();
#pragma unroll
  for (int s = 0; s < NSLOT; ++s)
    hbuf[s][t] = (smid[s][t] - mu8[s]) * rs8[s] * lg[t] + lb[t];
  __syncthreads();
  float a1[8] = {};
  for (int d = 0; d < DDIM; ++d) {
    float w = W1[(size_t)t * DDIM + d];
#pragma unroll
    for (int s = 0; s < NSLOT; ++s) a1[s] += hbuf[s][d] * w;
  }
#pragma unroll
  for (int s = 0; s < NSLOT; ++s) actv[s][t] = fmaxf(a1[s] + b1[t], 0.f);
  __syncthreads();
  float a2[8] = {};
  for (int d = 0; d < HID; ++d) {
    float w = W2[(size_t)t * HID + d];
#pragma unroll
    for (int s = 0; s < NSLOT; ++s) a2[s] += actv[s][d] * w;
  }
#pragma unroll
  for (int s = 0; s < NSLOT; ++s) {
    float o = smid[s][t] + a2[s] + b2[t];
    slots_out[((size_t)bat * NSLOT + s) * DDIM + t] = o;
    if (out2) out2[((size_t)bat * NSLOT + s) * DDIM + t] = o;
  }
}

// ------------------------------------------------------------------- launcher
extern "C" void kernel_launch(void* const* d_in, const int* in_sizes, int n_in,
                              void* d_out, int out_size, void* d_ws, size_t ws_size,
                              hipStream_t stream) {
  const float* x       = (const float*)d_in[0];
  const float* ln_in_g = (const float*)d_in[1];
  const float* ln_in_b = (const float*)d_in[2];
  const float* ln_s_g  = (const float*)d_in[3];
  const float* ln_s_b  = (const float*)d_in[4];
  const float* ln_ff_g = (const float*)d_in[5];
  const float* ln_ff_b = (const float*)d_in[6];
  const float* Wq      = (const float*)d_in[7];
  const float* bq      = (const float*)d_in[8];
  const float* Wk      = (const float*)d_in[9];
  const float* bk      = (const float*)d_in[10];
  const float* Wv      = (const float*)d_in[11];
  const float* bv      = (const float*)d_in[12];
  const float* W_ih    = (const float*)d_in[13];
  const float* b_ih    = (const float*)d_in[14];
  const float* W_hh    = (const float*)d_in[15];
  const float* b_hh    = (const float*)d_in[16];
  const float* W1      = (const float*)d_in[17];
  const float* b1      = (const float*)d_in[18];
  const float* W2      = (const float*)d_in[19];
  const float* b2      = (const float*)d_in[20];
  const float* emb     = (const float*)d_in[21];

  char* ws = (char*)d_ws;
  size_t off = 0;
  auto alloc = [&](size_t bytes) -> void* {
    void* p = ws + off;
    off += (bytes + 255) & ~(size_t)255;
    return p;
  };
  unsigned short* Kbuf = (unsigned short*)alloc((size_t)BATCH * NTOK * DDIM * 2);  // 64 MB
  unsigned short* VT   = (unsigned short*)alloc((size_t)BATCH * DDIM * NTOK * 2);  // 64 MB
  unsigned short* attn = (unsigned short*)alloc((size_t)BATCH * 16 * NTOK * 2);    //  4 MB
  unsigned short* qbf  = (unsigned short*)alloc((size_t)BATCH * 16 * DDIM * 2);
  float* sums  = (float*)alloc((size_t)BATCH * 16 * 4);
  float* slots = (float*)alloc((size_t)BATCH * NSLOT * DDIM * 4);
  float* upd   = (float*)alloc((size_t)BATCH * NSLOT * DDIM * 4);
  unsigned short* Wkb = (unsigned short*)alloc((size_t)DDIM * DDIM * 2);
  unsigned short* Wvb = (unsigned short*)alloc((size_t)DDIM * DDIM * 2);

  prep_w_kernel<<<dim3(DDIM, 2), DDIM, 0, stream>>>(Wk, Wv, Wkb, Wvb);
  init_slots_kernel<<<BATCH * NSLOT, DDIM, 0, stream>>>(emb, slots);
  ln_kv_kernel<<<(BATCH * NTOK / 16) / 4, 128, 0, stream>>>(
      x, ln_in_g, ln_in_b, Wkb, Wvb, bk, bv, Kbuf, VT);
  for (int it = 0; it < ITERS; ++it) {
    slot_q_kernel<<<BATCH, 256, 0, stream>>>(slots, ln_s_g, ln_s_b, Wq, bq, qbf, sums);
    dots_attn_kernel<<<(BATCH * (NTOK / 16)) / 4, 128, 0, stream>>>(Kbuf, qbf, attn, sums);
    updates_kernel<<<(BATCH * (DDIM / 16)) / 4, 128, 0, stream>>>(attn, VT, sums, upd);
    gru_mlp_kernel<<<BATCH, 256, 0, stream>>>(
        slots, upd, W_ih, b_ih, W_hh, b_hh, ln_ff_g, ln_ff_b, W1, b1, W2, b2,
        slots, (it == ITERS - 1) ? (float*)d_out : nullptr);
  }
  (void)in_sizes; (void)n_in; (void)out_size; (void)ws_size;
}
